// GaussianRenderer_34986803593229
// MI455X (gfx1250) — matrix operand order repaired and checked by reading, not compiled, not run on hardware
//
#include <hip/hip_runtime.h>
#include <hip/hip_fp16.h>

typedef __attribute__((ext_vector_type(16))) _Float16 v16h;
typedef __attribute__((ext_vector_type(8)))  float    v8f;
typedef __attribute__((ext_vector_type(4)))  float    v4f;

#define GS_H 200
#define GS_W 200
#define GS_P (GS_H * GS_W)       // 40000 pixels
#define GS_N 1024                // gaussians per batch
#define GS_B 2
#define GS_D 32                  // feature channels
#define REF_CHUNK 128            // reference early-stop granularity
#define KSTEP 32                 // gaussians per WMMA step (K of 16x16x32)
#define THRESH   0.05f
#define EPS_T    1e-4f
#define A_MIN    (1.0f / 255.0f)

// 32-byte packed per-gaussian params -> uniform reads become s_load_b256
struct GParam { float u, v, ca, cb, cc, op, p0, p1; };

// ---------------------------------------------------------------------------
// Kernel 1: projection + conic + prune, and features -> f16 transposed
// ---------------------------------------------------------------------------
__global__ void gs_prep(const float* __restrict__ feat,   // (B,N,32) f32
                        const float* __restrict__ means,  // (B,N,3)
                        const float* __restrict__ cov,    // (B,N,6)
                        const float* __restrict__ opac,   // (B,N,1)
                        GParam* __restrict__ params,      // (B*N) x 32B
                        _Float16* __restrict__ colsT) {   // (B,32,N) f16
  int i = blockIdx.x * blockDim.x + threadIdx.x;
  if (i >= GS_B * GS_N) return;
  int b = i >> 10, g = i & (GS_N - 1);

  const float* m  = means + i * 3;
  const float* c6 = cov   + i * 6;

  float u = -2.0f * m[1] + 100.0f;        // -SH*y + H/2
  float v = -2.0f * m[0] + 100.0f;        // -SW*x + W/2
  float a  = 4.0f * c6[3] + 0.3f;         // SH^2*Syy + 0.3
  float c  = 4.0f * c6[0] + 0.3f;         // SW^2*Sxx + 0.3
  float bb = 4.0f * c6[1];                // SH*SW*Sxy
  float det = a * c - bb * bb;
  bool valid = det > 0.0f;
  float ds = valid ? det : 1.0f;

  GParam q;
  q.u = u; q.v = v;
  q.ca = c / ds; q.cb = bb / ds; q.cc = a / ds;
  float o = opac[i];
  q.op = (o > THRESH && valid) ? o : 0.0f;
  q.p0 = 0.0f; q.p1 = 0.0f;
  params[i] = q;

  // features (g, ch) f32 -> colsT[b][ch][g] f16 (B-fragment friendly)
  const float* f = feat + i * GS_D;
  _Float16* ct = colsT + (size_t)b * GS_D * GS_N + g;
#pragma unroll
  for (int ch = 0; ch < GS_D; ++ch)
    ct[(size_t)ch * GS_N] = (_Float16)f[ch];
}

// ---------------------------------------------------------------------------
// Kernel 2: mean_count = mean over batches of #(opac > THRESH)
// ---------------------------------------------------------------------------
__global__ void gs_count(const float* __restrict__ opac, float* __restrict__ out) {
  __shared__ float sred[256];
  int t = threadIdx.x;
  float local = 0.0f;
  for (int i = t; i < GS_B * GS_N; i += 256)
    local += (opac[i] > THRESH) ? 1.0f : 0.0f;
  sred[t] = local;
  __syncthreads();
  for (int s = 128; s > 0; s >>= 1) {
    if (t < s) sred[t] += sred[t + s];
    __syncthreads();
  }
  if (t == 0) *out = sred[0] * (1.0f / (float)GS_B);
}

// ---------------------------------------------------------------------------
// Kernel 3: compositing + WMMA feature accumulation.
// One wave (32 lanes) per block; lane = one pixel. Per 32-gaussian step the
// lane runs the sequential transmittance recurrence (reference 128-chunk
// early-stop semantics), stages f16 weights in LDS, then 4x
// v_wmma_f32_16x16x32_f16: (16 px x 32 g) * (32 g x 16 ch) += (16x16 f32).
// ---------------------------------------------------------------------------
__global__ __launch_bounds__(32)
void gs_splat(const GParam* __restrict__ params,
              const _Float16* __restrict__ colsT,
              float* __restrict__ out) {
  const int lane  = threadIdx.x;           // 0..31
  const int b     = blockIdx.x / (GS_P / 32);
  const int tile  = blockIdx.x % (GS_P / 32);
  const int pbase = tile * 32;
  const int p     = pbase + lane;
  const float pxf = (float)(p % GS_W);
  const float pyf = (float)(p / GS_W);

  const GParam* __restrict__ gp = params + (size_t)b * GS_N;
  const _Float16* __restrict__ cb = colsT + (size_t)b * GS_D * GS_N;

  __shared__ _Float16 lds_w[32 * KSTEP];   // 2 KB: [pixel-lane][k]

  v8f acc00 = {}, acc01 = {}, acc10 = {}, acc11 = {};

  // per-pixel compositing state (reference 128-chunk semantics)
  float Tc = 1.0f;      // T at start of current 128-group
  float Tl = 1.0f;      // product of (1 - alpha_eff) within group
  float tr = 1.0f;      // product of (1 - alpha) (pre-early-stop) within group
  bool crossed = false;

  const int hs  = lane >> 4;               // lane half (K split selector)
  const int lid = lane & 15;

  for (int g0 = 0; g0 < GS_N; g0 += KSTEP) {
    // ---- sequential weight computation: this lane's pixel, 32 gaussians ----
#pragma unroll 8
    for (int k = 0; k < KSTEP; ++k) {
      GParam q = gp[g0 + k];               // uniform address -> scalar loads
      float dx = q.u - pxf;
      float dy = q.v - pyf;
      float power = -0.5f * (q.ca * dx * dx + q.cc * dy * dy) + q.cb * dx * dy;
      float alpha = fminf(0.99f, q.op * __expf(fminf(power, 0.0f)));
      if (power > 0.0f || alpha < A_MIN) alpha = 0.0f;
      tr *= (1.0f - alpha);                // un-zeroed running product
      if (Tc * tr < EPS_T) crossed = true; // reference T_incl test
      float aeff = crossed ? 0.0f : alpha;
      float w = aeff * Tl * Tc;            // alpha * T_excl * T
      Tl *= (1.0f - aeff);
      lds_w[lane * KSTEP + k] = (_Float16)w;
    }
    __syncthreads();

    // ---- B fragments: colsT[ch][g0 + hs*16 .. +15], N = channel = lane&15 ----
    union VU { v16h v; v4f q[2]; };
    VU b0, b1, a0, a1;
    {
      const _Float16* c0 = cb + (size_t)(lid)      * GS_N + g0 + hs * 16;
      const _Float16* c1 = cb + (size_t)(16 + lid) * GS_N + g0 + hs * 16;
      b0.q[0] = *(const v4f*)(c0);
      b0.q[1] = *(const v4f*)(c0 + 8);
      b1.q[0] = *(const v4f*)(c1);
      b1.q[1] = *(const v4f*)(c1 + 8);
    }
    // ---- A fragments from LDS: 16-bit A layout (contiguous K, matches B):
    //  VGPR0..3 <- K = hs*16 .. hs*16+7 ; VGPR4..7 <- K = hs*16+8 .. hs*16+15
    {
      const _Float16* r0 = &lds_w[(lid)      * KSTEP];
      const _Float16* r1 = &lds_w[(16 + lid) * KSTEP];
      a0.q[0] = *(const v4f*)(r0 + hs * 16);
      a0.q[1] = *(const v4f*)(r0 + hs * 16 + 8);
      a1.q[0] = *(const v4f*)(r1 + hs * 16);
      a1.q[1] = *(const v4f*)(r1 + hs * 16 + 8);
    }

    acc00 = __builtin_amdgcn_wmma_f32_16x16x32_f16(false, a0.v, false, b0.v,
                                                   (short)0, acc00, false, false);
    acc01 = __builtin_amdgcn_wmma_f32_16x16x32_f16(false, a0.v, false, b1.v,
                                                   (short)0, acc01, false, false);
    acc10 = __builtin_amdgcn_wmma_f32_16x16x32_f16(false, a1.v, false, b0.v,
                                                   (short)0, acc10, false, false);
    acc11 = __builtin_amdgcn_wmma_f32_16x16x32_f16(false, a1.v, false, b1.v,
                                                   (short)0, acc11, false, false);
    __syncthreads();                        // protect lds_w before next rewrite

    // ---- reference CHUNK=128 boundary: restart early-stop group ----
    if (((g0 + KSTEP) & (REF_CHUNK - 1)) == 0) {
      Tc *= Tl; Tl = 1.0f; tr = 1.0f; crossed = false;
    }
  }

  // ---- store: C/D layout VGPR r -> M = r + 8*hs, N = lid. out[b][ch][p] ----
  float* ob = out + (size_t)b * GS_D * GS_P;
  {
    float* o = ob + (size_t)(lid)      * GS_P + pbase + hs * 8;       // mt0,nt0
    *(v4f*)(o)     = (v4f){acc00[0], acc00[1], acc00[2], acc00[3]};
    *(v4f*)(o + 4) = (v4f){acc00[4], acc00[5], acc00[6], acc00[7]};
  }
  {
    float* o = ob + (size_t)(16 + lid) * GS_P + pbase + hs * 8;       // mt0,nt1
    *(v4f*)(o)     = (v4f){acc01[0], acc01[1], acc01[2], acc01[3]};
    *(v4f*)(o + 4) = (v4f){acc01[4], acc01[5], acc01[6], acc01[7]};
  }
  {
    float* o = ob + (size_t)(lid)      * GS_P + pbase + 16 + hs * 8;  // mt1,nt0
    *(v4f*)(o)     = (v4f){acc10[0], acc10[1], acc10[2], acc10[3]};
    *(v4f*)(o + 4) = (v4f){acc10[4], acc10[5], acc10[6], acc10[7]};
  }
  {
    float* o = ob + (size_t)(16 + lid) * GS_P + pbase + 16 + hs * 8;  // mt1,nt1
    *(v4f*)(o)     = (v4f){acc11[0], acc11[1], acc11[2], acc11[3]};
    *(v4f*)(o + 4) = (v4f){acc11[4], acc11[5], acc11[6], acc11[7]};
  }
}

// ---------------------------------------------------------------------------
extern "C" void kernel_launch(void* const* d_in, const int* in_sizes, int n_in,
                              void* d_out, int out_size, void* d_ws, size_t ws_size,
                              hipStream_t stream) {
  const float* feat  = (const float*)d_in[0];   // (B,N,32)
  const float* means = (const float*)d_in[1];   // (B,N,3)
  const float* cov   = (const float*)d_in[2];   // (B,N,6)
  const float* opac  = (const float*)d_in[3];   // (B,N,1)
  float* out = (float*)d_out;                   // (B,32,200,200) + mean_count

  // workspace: params (B*N*32B = 64KB) then colsT f16 (B*32*N*2B = 128KB)
  GParam*   params = (GParam*)d_ws;
  _Float16* colsT  = (_Float16*)((char*)d_ws + (size_t)GS_B * GS_N * sizeof(GParam));

  gs_prep<<<(GS_B * GS_N + 255) / 256, 256, 0, stream>>>(feat, means, cov, opac,
                                                         params, colsT);
  gs_count<<<1, 256, 0, stream>>>(opac, out + (size_t)GS_B * GS_D * GS_P);
  gs_splat<<<GS_B * (GS_P / 32), 32, 0, stream>>>(params, colsT, out);
}
